// GraphConvBlock_69621419868373
// MI455X (gfx1250) — compile-verified, matrix operand order checked by reading, not run on hardware
//
#include <hip/hip_runtime.h>
#include <hip/hip_bf16.h>
#include <stdint.h>

// ---------------- problem constants (match reference) ----------------
#define B_    2
#define C_    64
#define N_    8192
#define K_    32
#define HID_  128
#define OUT_  128
#define M_    (B_ * N_ * K_)     // 524288 edge rows
#define EPS_  1e-5f
#define SLOPE_ 0.2f

typedef __attribute__((ext_vector_type(2)))  float  v2f;
typedef __attribute__((ext_vector_type(8)))  float  v8f;
typedef __attribute__((ext_vector_type(8)))  __bf16 v8bf;
typedef __attribute__((ext_vector_type(16))) __bf16 v16bf;

// Build a 16-element bf16 WMMA operand from two contiguous 16B LDS runs.
// Per the CDNA5 16-bit A/B layout: lanes 0-15 hold K=[0..7]+[16..23],
// lanes 16-31 hold K=[8..15]+[24..31] within a 32-wide K chunk.
__device__ __forceinline__ v16bf ld_op16(const __bf16* chunk32, int hi8) {
  v8bf lo = *(const v8bf*)(chunk32 + hi8);
  v8bf hi = *(const v8bf*)(chunk32 + hi8 + 16);
  return __builtin_shufflevector(lo, hi, 0,1,2,3,4,5,6,7,8,9,10,11,12,13,14,15);
}

// CDNA5 async copy global -> LDS (tracked by ASYNCcnt)
__device__ __forceinline__ void async_b128(uint32_t lds_off, uint64_t gaddr) {
  asm volatile("global_load_async_to_lds_b128 %0, %1, off"
               :: "v"(lds_off), "v"(gaddr) : "memory");
}
__device__ __forceinline__ void wait_async0() {
  asm volatile("s_wait_asynccnt 0" ::: "memory");
}

// ---------------- kernel 0: transpose x -> xt, W -> bf16, zero stats ----------------
__global__ void init_kernel(const float* __restrict__ x,
                            const float* __restrict__ W1,
                            const float* __restrict__ W2,
                            float* __restrict__ xt,
                            __bf16* __restrict__ W1bf,
                            __bf16* __restrict__ W2bf,
                            float* __restrict__ stats) {
  int tid = blockIdx.x * 256 + threadIdx.x;
  if (tid < B_ * C_ * N_) {
    int b   = tid / (C_ * N_);
    int rem = tid % (C_ * N_);
    int c   = rem / N_;
    int n   = rem % N_;
    xt[((size_t)b * N_ + n) * C_ + c] = x[tid];
  }
  if (tid < HID_ * 2 * C_) W1bf[tid] = (__bf16)W1[tid];
  if (tid < OUT_ * HID_)   W2bf[tid] = (__bf16)W2[tid];
  if (tid < 512)           stats[tid] = 0.0f;   // sum1,sumsq1,sum2,sumsq2
}

// ---------------- kernel 1: kNN (f32 WMMA Gram tiles + top-33 selection) ----------------
__global__ void knn_kernel(const float* __restrict__ xt, int* __restrict__ idxOut) {
  __shared__ __align__(16) float rowF[32][C_];
  __shared__ __align__(16) float colF[128][C_];
  __shared__ float d2t[32][128];
  __shared__ float rowSq[32];
  __shared__ float colSq[128];

  const int t      = threadIdx.x;
  const int wgPerB = N_ / 32;                 // 256
  const int b      = blockIdx.x / wgPerB;
  const int rbase  = (blockIdx.x % wgPerB) * 32;
  const float* xtb = xt + (size_t)b * N_ * C_;

  for (int e = t; e < 32 * C_; e += 256) {
    int r = e >> 6, c = e & 63;
    rowF[r][c] = xtb[(size_t)(rbase + r) * C_ + c];
  }
  __syncthreads();
  if (t < 32) {
    float s = 0.0f;
    for (int c = 0; c < C_; ++c) { float f = rowF[t][c]; s += f * f; }
    rowSq[t] = s;
  }

  float bestd[33]; int besti[33];
  if (t < 32) for (int i = 0; i < 33; ++i) { bestd[i] = __builtin_inff(); besti[i] = 0; }

  const int w = t >> 5, l = t & 31;
  const int ln = l & 15, lhi = l >> 4;

  for (int c0 = 0; c0 < N_; c0 += 128) {
    if (t < 128) {                               // one column per thread (all from L2)
      float s = 0.0f;
      const float* src = xtb + (size_t)(c0 + t) * C_;
      #pragma unroll
      for (int c = 0; c < C_; c += 4) {
        float4 f = *(const float4*)(src + c);
        colF[t][c + 0] = f.x; colF[t][c + 1] = f.y;
        colF[t][c + 2] = f.z; colF[t][c + 3] = f.w;
        s += f.x * f.x + f.y * f.y + f.z * f.z + f.w * f.w;
      }
      colSq[t] = s;
    }
    __syncthreads();

    // 2 row-tiles x 8 col-tiles of 16x16; 8 waves -> 2 tiles each
    const int rt = w & 1;
    #pragma unroll
    for (int cc = 0; cc < 2; ++cc) {
      const int ct = (w >> 1) + cc * 4;
      v8f acc = {0.f,0.f,0.f,0.f,0.f,0.f,0.f,0.f};
      #pragma unroll
      for (int k = 0; k < C_; k += 4) {
        v2f a  = *(const v2f*)&rowF[rt * 16 + ln][k + 2 * lhi];
        v2f bm = *(const v2f*)&colF[ct * 16 + ln][k + 2 * lhi];
        acc = __builtin_amdgcn_wmma_f32_16x16x4_f32(false, a, false, bm,
                                                    (short)0, acc, false, false);
      }
      #pragma unroll
      for (int vv = 0; vv < 8; ++vv) {
        int rr = rt * 16 + vv + 8 * lhi;
        int jc = ct * 16 + ln;
        d2t[rr][jc] = rowSq[rr] + colSq[jc] - 2.0f * acc[vv];
      }
    }
    __syncthreads();

    if (t < 32) {                                // per-row top-33 insertion
      for (int j = 0; j < 128; ++j) {
        float d = d2t[t][j];
        if (d < bestd[32]) {
          int p = 32;
          while (p > 0 && bestd[p - 1] > d) {
            bestd[p] = bestd[p - 1]; besti[p] = besti[p - 1]; --p;
          }
          bestd[p] = d; besti[p] = c0 + j;
        }
      }
    }
    __syncthreads();
  }

  if (t < 32) {                                  // drop self (entry 0), keep 32 neighbors
    size_t o = ((size_t)b * N_ + rbase + t) * K_;
    for (int i = 0; i < K_; ++i) idxOut[o + i] = besti[i + 1];
  }
}

// ---------------- kernel 2: edge features + GEMM1 (bf16 WMMA) + BN1 stats ----------------
__global__ void edge_gemm1_kernel(const float* __restrict__ xt,
                                  const int* __restrict__ idx,
                                  const __bf16* __restrict__ Wbf,
                                  __bf16* __restrict__ yws,
                                  float* __restrict__ sum,
                                  float* __restrict__ sumsq) {
  __shared__ __align__(16) __bf16 Atile[128][128];   // edge feats, later y1 staging
  __shared__ __align__(16) __bf16 Wl[128][128];
  const int t  = threadIdx.x;
  const int m0 = blockIdx.x * 128;

  {  // async-stream W1(bf16) into LDS (2048 x 16B chunks, ASYNCcnt path)
    uint32_t ldsBase = (uint32_t)(uintptr_t)&Wl[0][0];
    #pragma unroll
    for (int i = 0; i < 8; ++i) {
      int e = i * 256 + t;
      async_b128(ldsBase + (uint32_t)e * 16u, (uint64_t)(uintptr_t)(Wbf + (size_t)e * 8));
    }
  }

  {  // gather + build 128x128 bf16 edge tile: [feats-center | center]
    const int r = t >> 1, half = t & 1;
    const int m = m0 + r;
    const int b = m / (N_ * K_);
    const int n = (m / K_) % N_;
    const int j = idx[m];
    const float* xtb = xt + (size_t)b * N_ * C_;
    const float* cn  = xtb + (size_t)n * C_;
    if (half == 0) {
      const float* cj = xtb + (size_t)j * C_;
      #pragma unroll
      for (int c = 0; c < C_; c += 4) {
        float4 fj = *(const float4*)(cj + c);
        float4 fn = *(const float4*)(cn + c);
        Atile[r][c + 0] = (__bf16)(fj.x - fn.x);
        Atile[r][c + 1] = (__bf16)(fj.y - fn.y);
        Atile[r][c + 2] = (__bf16)(fj.z - fn.z);
        Atile[r][c + 3] = (__bf16)(fj.w - fn.w);
      }
    } else {
      #pragma unroll
      for (int c = 0; c < C_; c += 4) {
        float4 fn = *(const float4*)(cn + c);
        Atile[r][64 + c + 0] = (__bf16)fn.x;
        Atile[r][64 + c + 1] = (__bf16)fn.y;
        Atile[r][64 + c + 2] = (__bf16)fn.z;
        Atile[r][64 + c + 3] = (__bf16)fn.w;
      }
    }
  }
  wait_async0();
  __syncthreads();

  // GEMM: each wave -> 16 rows x 128 cols, K=128 in 4 steps of 32
  const int w = t >> 5, l = t & 31;
  const int ln = l & 15, lhi = l >> 4, hi8 = lhi * 8;
  const int mr = w * 16;
  v8f acc[8];
  #pragma unroll
  for (int ot = 0; ot < 8; ++ot) acc[ot] = (v8f){0.f,0.f,0.f,0.f,0.f,0.f,0.f,0.f};
  #pragma unroll
  for (int ks = 0; ks < 4; ++ks) {
    const int kb = ks * 32;
    v16bf a = ld_op16(&Atile[mr + ln][kb], hi8);
    #pragma unroll
    for (int ot = 0; ot < 8; ++ot) {
      v16bf bm = ld_op16(&Wl[ot * 16 + ln][kb], hi8);
      acc[ot] = __builtin_amdgcn_wmma_f32_16x16x32_bf16(false, a, false, bm,
                                                        (short)0, acc[ot], false, false);
    }
  }
  // stage y1 (bf16) back into own rows of Atile (no cross-wave hazard)
  #pragma unroll
  for (int ot = 0; ot < 8; ++ot)
    #pragma unroll
    for (int vv = 0; vv < 8; ++vv)
      Atile[mr + vv + 8 * lhi][ot * 16 + ln] = (__bf16)acc[ot][vv];
  __syncthreads();

  {  // BN stats: 64x LDS pre-reduction, then one f32 atomic pair per (col,half)
    const int col = t & 127, rh = t >> 7;
    float s = 0.0f, s2 = 0.0f;
    for (int r = rh * 64; r < rh * 64 + 64; ++r) {
      float f = (float)Atile[r][col];
      s += f; s2 += f * f;
    }
    atomicAdd(&sum[col], s);
    atomicAdd(&sumsq[col], s2);
  }
  {  // coalesced b128 store of bf16 y1 tile (L2-resident working set)
    __bf16* yb = yws + (size_t)m0 * 128;
    const __bf16* flat = &Atile[0][0];
    #pragma unroll
    for (int i = 0; i < 8; ++i) {
      size_t off = (size_t)(i * 256 + t) * 8;
      *(v8bf*)(yb + off) = *(const v8bf*)(flat + off);
    }
  }
}

// ---------------- kernel 3/5: fold BN stats into scale/shift ----------------
__global__ void finalize_kernel(const float* __restrict__ sum,
                                const float* __restrict__ sumsq,
                                const float* __restrict__ g,
                                const float* __restrict__ bb,
                                float* __restrict__ scale,
                                float* __restrict__ shift) {
  int c = threadIdx.x;
  float cnt = (float)M_;
  float mu  = sum[c] / cnt;
  float var = sumsq[c] / cnt - mu * mu;
  float s   = g[c] * rsqrtf(var + EPS_);
  scale[c] = s;
  shift[c] = bb[c] - mu * s;
}

// ---------------- kernel 4: BN1+LReLU -> GEMM2 (in place) + BN2 stats ----------------
__global__ void gemm2_kernel(__bf16* __restrict__ yws,
                             const __bf16* __restrict__ Wbf,
                             const float* __restrict__ scale,
                             const float* __restrict__ shift,
                             float* __restrict__ sum,
                             float* __restrict__ sumsq) {
  __shared__ __align__(16) __bf16 Atile[128][128];
  __shared__ __align__(16) __bf16 Wl[128][128];
  __shared__ float sc[128], sh[128];
  const int t  = threadIdx.x;
  const int m0 = blockIdx.x * 128;

  {  // async-stream W2(bf16) into LDS
    uint32_t ldsBase = (uint32_t)(uintptr_t)&Wl[0][0];
    #pragma unroll
    for (int i = 0; i < 8; ++i) {
      int e = i * 256 + t;
      async_b128(ldsBase + (uint32_t)e * 16u, (uint64_t)(uintptr_t)(Wbf + (size_t)e * 8));
    }
  }
  if (t < 128) { sc[t] = scale[t]; sh[t] = shift[t]; }
  __syncthreads();

  __bf16* yb = yws + (size_t)m0 * 128;
  {  // load y1 tile, apply BN1 + LeakyReLU in f32, stage bf16 h1
    __bf16* flat = &Atile[0][0];
    #pragma unroll
    for (int i = 0; i < 8; ++i) {
      size_t off = (size_t)(i * 256 + t) * 8;
      v8bf v = *(const v8bf*)(yb + off);
      int colb = (int)(off & 127);
      #pragma unroll
      for (int q = 0; q < 8; ++q) {
        float f = sc[colb + q] * (float)v[q] + sh[colb + q];
        f = (f >= 0.0f) ? f : SLOPE_ * f;
        flat[off + q] = (__bf16)f;
      }
    }
  }
  wait_async0();
  __syncthreads();

  const int w = t >> 5, l = t & 31;
  const int ln = l & 15, lhi = l >> 4, hi8 = lhi * 8;
  const int mr = w * 16;
  v8f acc[8];
  #pragma unroll
  for (int ot = 0; ot < 8; ++ot) acc[ot] = (v8f){0.f,0.f,0.f,0.f,0.f,0.f,0.f,0.f};
  #pragma unroll
  for (int ks = 0; ks < 4; ++ks) {
    const int kb = ks * 32;
    v16bf a = ld_op16(&Atile[mr + ln][kb], hi8);
    #pragma unroll
    for (int ot = 0; ot < 8; ++ot) {
      v16bf bm = ld_op16(&Wl[ot * 16 + ln][kb], hi8);
      acc[ot] = __builtin_amdgcn_wmma_f32_16x16x32_bf16(false, a, false, bm,
                                                        (short)0, acc[ot], false, false);
    }
  }
  #pragma unroll
  for (int ot = 0; ot < 8; ++ot)
    #pragma unroll
    for (int vv = 0; vv < 8; ++vv)
      Atile[mr + vv + 8 * lhi][ot * 16 + ln] = (__bf16)acc[ot][vv];
  __syncthreads();

  {
    const int col = t & 127, rh = t >> 7;
    float s = 0.0f, s2 = 0.0f;
    for (int r = rh * 64; r < rh * 64 + 64; ++r) {
      float f = (float)Atile[r][col];
      s += f; s2 += f * f;
    }
    atomicAdd(&sum[col], s);
    atomicAdd(&sumsq[col], s2);
  }
  {  // overwrite same workspace tile with y2 (safe: fully read above)
    const __bf16* flat = &Atile[0][0];
    #pragma unroll
    for (int i = 0; i < 8; ++i) {
      size_t off = (size_t)(i * 256 + t) * 8;
      *(v8bf*)(yb + off) = *(const v8bf*)(flat + off);
    }
  }
}

// ---------------- kernel 6: BN2 + LReLU + max over K + transpose ----------------
__global__ void maxpool_kernel(const __bf16* __restrict__ yws,
                               const float* __restrict__ scale,
                               const float* __restrict__ shift,
                               float* __restrict__ out) {
  const int bn = blockIdx.x;
  const int o  = threadIdx.x;
  const int b  = bn >> 13;              // N_ = 8192
  const int n  = bn & (N_ - 1);
  const __bf16* base = yws + (size_t)bn * K_ * 128;
  const float s = scale[o], h = shift[o];
  float mx = -__builtin_inff();
  #pragma unroll 4
  for (int k = 0; k < K_; ++k) {
    float f = s * (float)base[(size_t)k * 128 + o] + h;
    f = (f >= 0.0f) ? f : SLOPE_ * f;
    mx = fmaxf(mx, f);
  }
  out[((size_t)b * OUT_ + o) * N_ + n] = mx;
}

// ---------------- host launcher ----------------
extern "C" void kernel_launch(void* const* d_in, const int* in_sizes, int n_in,
                              void* d_out, int out_size, void* d_ws, size_t ws_size,
                              hipStream_t stream) {
  (void)in_sizes; (void)n_in; (void)out_size; (void)ws_size;
  const float* x  = (const float*)d_in[0];
  // d_in[1] = k (scalar) — fixed at 32
  const float* W1 = (const float*)d_in[2];
  const float* g1 = (const float*)d_in[3];
  const float* b1 = (const float*)d_in[4];
  const float* W2 = (const float*)d_in[5];
  const float* g2 = (const float*)d_in[6];
  const float* b2 = (const float*)d_in[7];

  float* out    = (float*)d_out;
  int*   idxOut = (int*)((float*)d_out + (size_t)B_ * OUT_ * N_);

  // workspace layout
  char* ws = (char*)d_ws;
  const size_t XT_OFF    = 0;                                   // 4 MB f32
  const size_t W1BF_OFF  = XT_OFF + (size_t)B_ * N_ * C_ * 4;   // 32 KB bf16
  const size_t W2BF_OFF  = W1BF_OFF + (size_t)HID_ * 2 * C_ * 2;
  const size_t STAT_OFF  = W2BF_OFF + (size_t)OUT_ * HID_ * 2;  // 4 KB f32
  const size_t YWS_OFF   = (STAT_OFF + 4096 + 255) & ~(size_t)255;  // 128 MB bf16

  float*  xt    = (float*)(ws + XT_OFF);
  __bf16* W1bf  = (__bf16*)(ws + W1BF_OFF);
  __bf16* W2bf  = (__bf16*)(ws + W2BF_OFF);
  float*  stats = (float*)(ws + STAT_OFF);
  float*  sum1   = stats;        float* sumsq1 = stats + 128;
  float*  sum2   = stats + 256;  float* sumsq2 = stats + 384;
  float*  scale1 = stats + 512;  float* shift1 = stats + 640;
  float*  scale2 = stats + 768;  float* shift2 = stats + 896;
  __bf16* yws   = (__bf16*)(ws + YWS_OFF);

  init_kernel<<<(B_ * C_ * N_ + 255) / 256, 256, 0, stream>>>(
      x, W1, W2, xt, W1bf, W2bf, stats);
  knn_kernel<<<B_ * (N_ / 32), 256, 0, stream>>>(xt, idxOut);
  edge_gemm1_kernel<<<M_ / 128, 256, 0, stream>>>(xt, idxOut, W1bf, yws, sum1, sumsq1);
  finalize_kernel<<<1, 128, 0, stream>>>(sum1, sumsq1, g1, b1, scale1, shift1);
  gemm2_kernel<<<M_ / 128, 256, 0, stream>>>(yws, W2bf, scale1, shift1, sum2, sumsq2);
  finalize_kernel<<<1, 128, 0, stream>>>(sum2, sumsq2, g2, b2, scale2, shift2);
  maxpool_kernel<<<B_ * N_, 128, 0, stream>>>(yws, scale2, shift2, out);
}
